// LSTMCHARTagger_34969623724081
// MI455X (gfx1250) — matrix operand order repaired and checked by reading, not compiled, not run on hardware
//
#include <hip/hip_runtime.h>
#include <hip/hip_bf16.h>
#include <math.h>

// ---------------------------------------------------------------------------
// LSTM char tagger for MI455X (gfx1250, wave32, WMMA 16x16x32 f16).
// S=4096, WC=12, CV=CE=CH=128, WV=100000, WE=256, HW=384, TAGS=48.
// ---------------------------------------------------------------------------

typedef __attribute__((ext_vector_type(16))) _Float16 v16h;
typedef __attribute__((ext_vector_type(2)))  _Float16 v2h;
typedef __attribute__((ext_vector_type(8)))  float    v8f;

__device__ __forceinline__ float sigf(float x) { return 1.0f / (1.0f + __expf(-x)); }

// Assemble one 16x32 f16 A-tile (M x K) from a row-major f16 LDS buffer into the
// CDNA5 WMMA A-operand lane layout (ISA 7.12.2, 16-bit A 16x32):
//   lanes 0-15 : M=lane,   halves j<8 -> K=k0+j,     j>=8 -> K=k0+16+(j-8)
//   lanes16-31 : M=lane-16, halves j<8 -> K=k0+8+j,  j>=8 -> K=k0+24+(j-8)
__device__ __forceinline__ v16h load_a16(const _Float16* src, int ldk, int k0, int lane) {
    int m  = lane & 15;
    int kb = k0 + ((lane >> 4) << 3);
    v16h a;
#pragma unroll
    for (int j = 0; j < 16; ++j) {
        int k = kb + ((j >> 3) << 4) + (j & 7);
        a[j] = src[m * ldk + k];
    }
    return a;
}

// ---------------------------------------------------------------------------
// Prep kernels
// ---------------------------------------------------------------------------

// Re-pack a row-major f32 weight W[N][K] (z = x @ W^T) into WMMA B-operand tiles:
// tile index = kt*(N/16)+nt; inside a tile each lane owns 16 contiguous halves:
//   n = nt*16 + (lane&15),  k = kt*32 + (lane>=16 ? 8 : 0) + (j>=8 ? 16 : 0) + (j&7)
__global__ void pack_b_f16(const float* __restrict__ W, _Float16* __restrict__ out,
                           int N, int K) {
    int idx = blockIdx.x * 256 + threadIdx.x;
    if (idx >= N * K) return;
    int j    = idx & 15;
    int lane = (idx >> 4) & 31;
    int tile = idx >> 9;
    int NT   = N >> 4;
    int nt   = tile % NT;
    int kt   = tile / NT;
    int n    = nt * 16 + (lane & 15);
    int k    = kt * 32 + ((lane >> 4) << 3) + ((j >> 3) << 4) + (j & 7);
    out[idx] = (_Float16)W[n * K + k];
}

// Whh[1536][384] -> k-major packed f16 pairs: out[kp*1536 + n] = {W[n][2kp], W[n][2kp+1]}
// Halves the per-step L2 traffic of the serial scan (1.18 MB/step instead of 2.36 MB).
__global__ void pack_whh_h2(const float* __restrict__ W, v2h* __restrict__ out) {
    int idx = blockIdx.x * 256 + threadIdx.x;   // 192*1536 elements
    if (idx >= 192 * 1536) return;
    int kp = idx / 1536;
    int n  = idx % 1536;
    v2h p;
    p.x = (_Float16)W[n * 384 + 2 * kp];
    p.y = (_Float16)W[n * 384 + 2 * kp + 1];
    out[idx] = p;
}

__global__ void add_bias(const float* __restrict__ a, const float* __restrict__ b,
                         float* __restrict__ out, int n) {
    int i = blockIdx.x * 256 + threadIdx.x;
    if (i < n) out[i] = a[i] + b[i];
}

// Fill word-embedding half of X: X[t][128..383] = word_emb[sentence[t]] (f32->f16)
__global__ void fill_word_emb(const int* __restrict__ sentence,
                              const float* __restrict__ word_emb,
                              _Float16* __restrict__ X) {
    int idx = blockIdx.x * 256 + threadIdx.x;   // 4096*256 elements
    int t = idx >> 8;
    int e = idx & 255;
    X[t * 384 + 128 + e] = (_Float16)word_emb[sentence[t] * 256 + e];
}

// ---------------------------------------------------------------------------
// Kernel 1: char LSTM. One wave per 16-word tile; 12 steps; all GEMM via WMMA.
// Per step per wave: 8 hidden-unit tiles x 4 gates x 8 K-chunks = 256 WMMAs.
// B-tiles are re-read from L2 every step (256 KB resident); an asm memory
// clobber inside the time loop stops LICM from hoisting 2048 VGPRs of B-tiles
// out of the loop and spilling them to scratch (seen in round-1 disasm).
// ---------------------------------------------------------------------------
__global__ void __launch_bounds__(128) char_lstm_kernel(
    const int* __restrict__ word_chars, const float* __restrict__ char_emb,
    const _Float16* __restrict__ WihP, const _Float16* __restrict__ WhhP,
    const float* __restrict__ bias, _Float16* __restrict__ X) {
    __shared__ _Float16 sx[4][16 * 128];
    __shared__ _Float16 sh[4][16 * 128];
    int lane = threadIdx.x & 31;
    int w    = threadIdx.x >> 5;
    int tile = blockIdx.x * 4 + w;
    _Float16* sX = sx[w];
    _Float16* sH = sh[w];

#pragma unroll
    for (int i = 0; i < 64; ++i) sH[lane * 64 + i] = (_Float16)0.0f;

    v8f c[8] = {};
    float bv[4][8];
#pragma unroll
    for (int g = 0; g < 4; ++g)
#pragma unroll
        for (int j = 0; j < 8; ++j) bv[g][j] = bias[(g * 8 + j) * 16 + (lane & 15)];

    __syncthreads();

    for (int t = 0; t < 12; ++t) {
        asm volatile("" ::: "memory");   // forbid hoisting loop-invariant B loads

        // stage gathered char embeddings (f32 -> f16) into LDS
        for (int m = 0; m < 16; ++m) {
            int ch = word_chars[(tile * 16 + m) * 12 + t];
#pragma unroll
            for (int q = 0; q < 4; ++q) {
                int e = lane + q * 32;
                sX[m * 128 + e] = (_Float16)char_emb[ch * 128 + e];
            }
        }
        __syncthreads();

        v16h ax[4], ah[4];
#pragma unroll
        for (int kt = 0; kt < 4; ++kt) {
            ax[kt] = load_a16(sX, 128, kt * 32, lane);
            ah[kt] = load_a16(sH, 128, kt * 32, lane);
        }
        __syncthreads();   // everyone has h in registers; safe to overwrite sH

#pragma unroll
        for (int j = 0; j < 8; ++j) {        // hidden-unit tile (16 units each)
            v8f zg[4];
#pragma unroll
            for (int g = 0; g < 4; ++g) {    // gates i,f,g,o
                int nt = g * 8 + j;
                v8f acc = {};
#pragma unroll
                for (int kt = 0; kt < 4; ++kt) {
                    v16h b = *(const v16h*)(WihP + (kt * 32 + nt) * 512 + lane * 16);
                    acc = __builtin_amdgcn_wmma_f32_16x16x32_f16(
                        false, ax[kt], false, b, (short)0, acc, false, false);
                }
#pragma unroll
                for (int kt = 0; kt < 4; ++kt) {
                    v16h b = *(const v16h*)(WhhP + (kt * 32 + nt) * 512 + lane * 16);
                    acc = __builtin_amdgcn_wmma_f32_16x16x32_f16(
                        false, ah[kt], false, b, (short)0, acc, false, false);
                }
#pragma unroll
                for (int r = 0; r < 8; ++r) acc[r] += bv[g][j];
                zg[g] = acc;
            }
#pragma unroll
            for (int r = 0; r < 8; ++r) {
                float ig = sigf(zg[0][r]);
                float fg = sigf(zg[1][r]);
                float gg = tanhf(zg[2][r]);
                float og = sigf(zg[3][r]);
                float cc = fg * c[j][r] + ig * gg;
                c[j][r]  = cc;
                float hv = og * tanhf(cc);
                int M = r + ((lane >> 4) << 3);   // D-matrix layout (ISA 7.12.2)
                int N = j * 16 + (lane & 15);
                sH[M * 128 + N] = (_Float16)hv;
            }
        }
        __syncthreads();
    }

    // final char hidden -> X[:, 0:128]
    for (int m = 0; m < 16; ++m) {
#pragma unroll
        for (int q = 0; q < 4; ++q) {
            int e = lane + q * 32;
            X[(tile * 16 + m) * 384 + e] = sH[m * 128 + e];
        }
    }
}

// ---------------------------------------------------------------------------
// Kernel 2a: PRE = X[4096x384] @ w_Wih^T[384x1536] + (bih+bhh).  WMMA GEMM.
// Block = 4 waves; block covers one M-tile and 4 N-tiles; 12 K-chunk WMMAs.
// ---------------------------------------------------------------------------
__global__ void __launch_bounds__(128) pre_gemm_kernel(
    const _Float16* __restrict__ X, const _Float16* __restrict__ WihP,
    const float* __restrict__ bias, float* __restrict__ PRE) {
    __shared__ _Float16 sA[16 * 384];
    int lane = threadIdx.x & 31;
    int w    = threadIdx.x >> 5;
    int mt   = blockIdx.x / 24;
    int ng   = blockIdx.x % 24;

    for (int i = threadIdx.x; i < 16 * 384; i += 128)
        sA[i] = X[(mt * 16 + (i / 384)) * 384 + (i % 384)];
    __syncthreads();

    int nt = ng * 4 + w;
    v8f acc = {};
#pragma unroll
    for (int kt = 0; kt < 12; ++kt) {
        v16h a = load_a16(sA, 384, kt * 32, lane);
        v16h b = *(const v16h*)(WihP + (kt * 96 + nt) * 512 + lane * 16);
        acc = __builtin_amdgcn_wmma_f32_16x16x32_f16(
            false, a, false, b, (short)0, acc, false, false);
    }
    float bvv = bias[nt * 16 + (lane & 15)];
#pragma unroll
    for (int r = 0; r < 8; ++r) {
        int M = r + ((lane >> 4) << 3);
        int N = lane & 15;
        PRE[(mt * 16 + M) * 1536 + nt * 16 + N] = acc[r] + bvv;
    }
}

// ---------------------------------------------------------------------------
// Kernel 2b: serial word-LSTM scan. 1 workgroup, 384 threads (1 per hidden
// unit, computing all 4 of its gates). h broadcast via LDS; Whh packed as
// k-major f16 pairs so each dword load feeds two k-steps for one gate and
// all loads coalesce; the 1.18 MB matrix stays L2-resident (192 MB L2).
// ---------------------------------------------------------------------------
__global__ void __launch_bounds__(384) word_lstm_scan(
    const float* __restrict__ PRE, const v2h* __restrict__ WhhP2,
    float* __restrict__ HS) {
    __shared__ float hS[384];
    int u = threadIdx.x;
    hS[u] = 0.0f;
    float c = 0.0f;
    __syncthreads();

    for (int t = 0; t < 4096; ++t) {
        const float* pre = PRE + t * 1536;
        float z0 = pre[u];
        float z1 = pre[384 + u];
        float z2 = pre[768 + u];
        float z3 = pre[1152 + u];
        if (t + 1 < 4096) {   // gfx1250 global_prefetch_b8 on next step's rows
            __builtin_prefetch(PRE + (t + 1) * 1536 + u, 0, 1);
            __builtin_prefetch(PRE + (t + 1) * 1536 + 768 + u, 0, 1);
        }
        for (int kp = 0; kp < 192; kp += 2) {
#pragma unroll
            for (int q = 0; q < 2; ++q) {
                int k2 = kp + q;
                float h0 = hS[2 * k2];
                float h1 = hS[2 * k2 + 1];
                const v2h* wr = WhhP2 + k2 * 1536 + u;
                v2h w0 = wr[0];
                v2h w1 = wr[384];
                v2h w2 = wr[768];
                v2h w3 = wr[1152];
                z0 = fmaf((float)w0.x, h0, fmaf((float)w0.y, h1, z0));
                z1 = fmaf((float)w1.x, h0, fmaf((float)w1.y, h1, z1));
                z2 = fmaf((float)w2.x, h0, fmaf((float)w2.y, h1, z2));
                z3 = fmaf((float)w3.x, h0, fmaf((float)w3.y, h1, z3));
            }
        }
        float ig = sigf(z0), fg = sigf(z1), gg = tanhf(z2), og = sigf(z3);
        c = fg * c + ig * gg;
        float h = og * tanhf(c);
        __syncthreads();      // all reads of old h done
        hS[u] = h;
        __syncthreads();      // new h visible
        HS[t * 384 + u] = h;
    }
}

// ---------------------------------------------------------------------------
// Kernel 3: tag_space = HS @ lin_W^T + lin_b, then log_softmax over 48 tags.
// One wave per sentence position; shuffle reductions (wave32).
// ---------------------------------------------------------------------------
__global__ void __launch_bounds__(256) tag_softmax_kernel(
    const float* __restrict__ HS, const float* __restrict__ linW,
    const float* __restrict__ linb, float* __restrict__ out) {
    __shared__ float row[8][384];
    int lane = threadIdx.x & 31;
    int w    = threadIdx.x >> 5;
    int r    = blockIdx.x * 8 + w;
    float* rw = row[w];
    for (int i = lane; i < 384; i += 32) rw[i] = HS[r * 384 + i];
    __syncthreads();

    int tag0 = lane;
    int tag1 = 32 + lane;                       // valid only for lane < 16
    float a0 = linb[tag0];
    float a1 = (lane < 16) ? linb[tag1] : 0.0f;
    for (int k = 0; k < 384; ++k) {
        float x = rw[k];
        a0 = fmaf(linW[tag0 * 384 + k], x, a0);
        if (lane < 16) a1 = fmaf(linW[tag1 * 384 + k], x, a1);
    }
    float m = fmaxf(a0, (lane < 16) ? a1 : -INFINITY);
#pragma unroll
    for (int off = 16; off; off >>= 1) m = fmaxf(m, __shfl_xor(m, off, 32));
    float s = __expf(a0 - m) + ((lane < 16) ? __expf(a1 - m) : 0.0f);
#pragma unroll
    for (int off = 16; off; off >>= 1) s += __shfl_xor(s, off, 32);
    float lse = m + __logf(s);
    out[r * 48 + tag0] = a0 - lse;
    if (lane < 16) out[r * 48 + tag1] = a1 - lse;
}

// ---------------------------------------------------------------------------
// Host launch
// ---------------------------------------------------------------------------
extern "C" void kernel_launch(void* const* d_in, const int* in_sizes, int n_in,
                              void* d_out, int out_size, void* d_ws, size_t ws_size,
                              hipStream_t stream) {
    const int*   word_chars = (const int*)d_in[0];
    const int*   sentence   = (const int*)d_in[1];
    const float* char_emb   = (const float*)d_in[2];
    const float* word_emb   = (const float*)d_in[3];
    const float* c_Wih      = (const float*)d_in[4];
    const float* c_Whh      = (const float*)d_in[5];
    const float* c_bih      = (const float*)d_in[6];
    const float* c_bhh      = (const float*)d_in[7];
    const float* w_Wih      = (const float*)d_in[8];
    const float* w_Whh      = (const float*)d_in[9];
    const float* w_bih      = (const float*)d_in[10];
    const float* w_bhh      = (const float*)d_in[11];
    const float* lin_W      = (const float*)d_in[12];
    const float* lin_b      = (const float*)d_in[13];
    float* out = (float*)d_out;

    char* ws = (char*)d_ws;
    size_t off = 0;
    auto alloc = [&](size_t bytes) {
        char* p = ws + off;
        off = (off + bytes + 255) & ~size_t(255);
        return p;
    };
    _Float16* cWihP = (_Float16*)alloc(512 * 128 * sizeof(_Float16));
    _Float16* cWhhP = (_Float16*)alloc(512 * 128 * sizeof(_Float16));
    _Float16* wWihP = (_Float16*)alloc(1536 * 384 * sizeof(_Float16));
    v2h*      WhhP2 = (v2h*)     alloc(192 * 1536 * sizeof(v2h));
    float*    cB    = (float*)   alloc(512 * sizeof(float));
    float*    wB    = (float*)   alloc(1536 * sizeof(float));
    _Float16* X     = (_Float16*)alloc(4096 * 384 * sizeof(_Float16));
    float*    PRE   = (float*)   alloc(4096 * 1536 * sizeof(float));
    float*    HS    = (float*)   alloc(4096 * 384 * sizeof(float));

    // ---- weight prep ----
    pack_b_f16<<<(512 * 128 + 255) / 256, 256, 0, stream>>>(c_Wih, cWihP, 512, 128);
    pack_b_f16<<<(512 * 128 + 255) / 256, 256, 0, stream>>>(c_Whh, cWhhP, 512, 128);
    pack_b_f16<<<(1536 * 384 + 255) / 256, 256, 0, stream>>>(w_Wih, wWihP, 1536, 384);
    pack_whh_h2<<<(192 * 1536 + 255) / 256, 256, 0, stream>>>(w_Whh, WhhP2);
    add_bias<<<2, 256, 0, stream>>>(c_bih, c_bhh, cB, 512);
    add_bias<<<6, 256, 0, stream>>>(w_bih, w_bhh, wB, 1536);
    fill_word_emb<<<4096, 256, 0, stream>>>(sentence, word_emb, X);

    // ---- char LSTM (WMMA), writes X[:, 0:128] ----
    char_lstm_kernel<<<64, 128, 0, stream>>>(word_chars, char_emb, cWihP, cWhhP, cB, X);

    // ---- hoisted input GEMM for word LSTM (WMMA) ----
    pre_gemm_kernel<<<256 * 24, 128, 0, stream>>>(X, wWihP, wB, PRE);

    // ---- serial word-LSTM scan ----
    word_lstm_scan<<<1, 384, 0, stream>>>(PRE, WhhP2, HS);

    // ---- classifier + log_softmax ----
    tag_softmax_kernel<<<512, 256, 0, stream>>>(HS, lin_W, lin_b, out);

    (void)in_sizes; (void)n_in; (void)out_size; (void)ws_size;
}
